// EdgeUpdaterSAM_1_20985210208289
// MI455X (gfx1250) — matrix operand order repaired and checked by reading, not compiled, not run on hardware
//
#include <hip/hip_runtime.h>

typedef __attribute__((ext_vector_type(16))) _Float16 v16h;
typedef __attribute__((ext_vector_type(8)))  _Float16 v8h;
typedef __attribute__((ext_vector_type(8)))  float    v8f;

#define N_NODES  768
#define NODE_DIM 256
#define HIDDEN   64
#define EDGE_DIM 128

// Intra-wave LDS ordering: DS ops from one wave complete in order; we only need
// the outstanding ds stores drained before cross-lane ds loads, plus a compiler
// scheduling fence.
__device__ __forceinline__ void wave_lds_sync() {
  __builtin_amdgcn_wave_barrier();
  asm volatile("s_wait_dscnt 0" ::: "memory");
  __builtin_amdgcn_wave_barrier();
}

// ---------------- prep 1: n_i = x@Wi+bi, n_j = x@Wj+bj ----------------
__global__ __launch_bounds__(128)
void prep_proj(const float* __restrict__ x,
               const float* __restrict__ Wi, const float* __restrict__ bi,
               const float* __restrict__ Wj, const float* __restrict__ bj,
               float* __restrict__ nI, float* __restrict__ nJ) {
  const int row = blockIdx.x;
  const int t = threadIdx.x;
  const float* xr = x + row * NODE_DIM;
  if (t < HIDDEN) {
    float acc = bi[t];
    for (int k = 0; k < NODE_DIM; ++k) acc += xr[k] * Wi[k * HIDDEN + t];
    nI[row * HIDDEN + t] = acc;
  } else {
    const int c = t - HIDDEN;
    float acc = bj[c];
    for (int k = 0; k < NODE_DIM; ++k) acc += xr[k] * Wj[k * HIDDEN + c];
    nJ[row * HIDDEN + c] = acc;
  }
}

// ---------------- prep 2: pack W1/W2 into f16 WMMA B-fragment order ----------
// B fragment (16x16x32 f16): lane l holds column N = l%16, K = 32*s + 16*(l/16) + j,
// j = 0..15 contiguous. Flat layout: ((s*NT + n)*32 + l)*16 + j.
__global__ __launch_bounds__(256)
void prep_pack(const float* __restrict__ W1, const float* __restrict__ W2,
               _Float16* __restrict__ w1p, _Float16* __restrict__ w2p) {
  for (int idx = threadIdx.x; idx < 12288; idx += 256) {
    if (idx < 4096) {                       // W1: 64x64, 2 K-steps x 4 N-tiles
      const int jj = idx & 15;
      const int l  = (idx >> 4) & 31;
      const int n  = (idx >> 9) & 3;
      const int s  = idx >> 11;
      const int k  = 32 * s + 16 * (l >> 4) + jj;
      const int col = 16 * n + (l & 15);
      w1p[idx] = (_Float16)W1[k * HIDDEN + col];
    } else {                                // W2: 64x128, 2 K-steps x 8 N-tiles
      const int id = idx - 4096;
      const int jj = id & 15;
      const int l  = (id >> 4) & 31;
      const int n  = (id >> 9) & 7;
      const int s  = id >> 12;
      const int k  = 32 * s + 16 * (l >> 4) + jj;
      const int col = 16 * n + (l & 15);
      w2p[id] = (_Float16)W2[k * EDGE_DIM + col];
    }
  }
}

// ---------------- main: pairwise LN -> GEMM1(relu) -> GEMM2 ----------------
__global__ __launch_bounds__(256)
void edge_main(const float* __restrict__ nI, const float* __restrict__ nJ,
               const _Float16* __restrict__ w1p, const _Float16* __restrict__ w2p,
               const float* __restrict__ ln_g, const float* __restrict__ ln_b,
               const float* __restrict__ b1, const float* __restrict__ b2,
               float* __restrict__ out) {
  __shared__ __align__(32) _Float16 sW1[4096];         // 8 KB
  __shared__ __align__(32) _Float16 sW2[8192];         // 16 KB
  __shared__ float sLN[320];                           // g(64) b(64) b1(64) b2(128)
  __shared__ __align__(32) _Float16 sA[8][16 * 64];    // per-wave LN'd A tile
  __shared__ __align__(32) _Float16 sH[8][16 * 64];    // per-wave relu tile

  const int tid = threadIdx.x;
  {
    const uint4* g1 = (const uint4*)w1p; uint4* d1 = (uint4*)sW1;
    for (int k = tid; k < 512; k += 256) d1[k] = g1[k];
    const uint4* g2 = (const uint4*)w2p; uint4* d2 = (uint4*)sW2;
    for (int k = tid; k < 1024; k += 256) d2[k] = g2[k];
    for (int k = tid; k < 320; k += 256) {
      float v;
      if (k < 64)       v = ln_g[k];
      else if (k < 128) v = ln_b[k - 64];
      else if (k < 192) v = b1[k - 128];
      else              v = b2[k - 192];
      sLN[k] = v;
    }
  }
  __syncthreads();

  const int w    = tid >> 5;
  const int lane = tid & 31;
  const int lr   = lane & 15;       // row within tile (also N col for C/D frag)
  const int hi   = lane >> 4;       // half-wave selector
  const int pbase = blockIdx.x * 128 + w * 16;   // 16 pairs per wave
  const int i  = (pbase % N_NODES) + lr;         // 768 % 16 == 0 -> no straddle
  const int j  = pbase / N_NODES;                // constant across the wave
  const int cb = hi * 32;                        // column block [cb, cb+32)

  // ---- h = n_i[i] + n_j[j]; accumulate LN stats over this lane's 32 cols ----
  float h[32];
  float s1 = 0.f, s2 = 0.f;
  const float* pi = nI + i * HIDDEN + cb;
  const float* pj = nJ + j * HIDDEN + cb;
#pragma unroll
  for (int c = 0; c < 32; c += 4) {
    float4 a = *(const float4*)(pi + c);
    float4 b = *(const float4*)(pj + c);
    float v0 = a.x + b.x, v1 = a.y + b.y, v2 = a.z + b.z, v3 = a.w + b.w;
    h[c] = v0; h[c + 1] = v1; h[c + 2] = v2; h[c + 3] = v3;
    s1 += v0 + v1 + v2 + v3;
    s2 += v0 * v0 + v1 * v1 + v2 * v2 + v3 * v3;
  }
  // partner lane (l ^ 16) holds the other 32 columns of the same row
  s1 += __shfl_xor(s1, 16);
  s2 += __shfl_xor(s2, 16);
  const float mean = s1 * (1.f / 64.f);
  const float rstd = rsqrtf(s2 * (1.f / 64.f) - mean * mean + 1e-5f);

  // ---- normalize, f16-convert, write A tile (row-major [16][64] halves) ----
  _Float16* pA = &sA[w][lr * 64 + cb];
#pragma unroll
  for (int ch = 0; ch < 4; ++ch) {
    v8h pk;
#pragma unroll
    for (int q = 0; q < 8; ++q) {
      const int c = ch * 8 + q;
      float v = (h[c] - mean) * rstd * sLN[cb + c] + sLN[64 + cb + c];
      pk[q] = (_Float16)v;
    }
    *(v8h*)(pA + ch * 8) = pk;
  }
  wave_lds_sync();

  // ---- GEMM1: [16x64] x [64x64] with K=32 f16 WMMA, bias + ReLU -> sH ----
  const _Float16* aBase = &sA[w][lr * 64];
#pragma unroll
  for (int n = 0; n < 4; ++n) {
    v8f acc = {0.f, 0.f, 0.f, 0.f, 0.f, 0.f, 0.f, 0.f};
#pragma unroll
    for (int s = 0; s < 2; ++s) {
      // A frag: lane holds K = 32s + 16*hi + {0..15} (matches packed B)
      v8h lo  = *(const v8h*)(aBase + 32 * s + 16 * hi);
      v8h hi8 = *(const v8h*)(aBase + 32 * s + 16 * hi + 8);
      v16h afrag = __builtin_shufflevector(lo, hi8, 0, 1, 2, 3, 4, 5, 6, 7,
                                           8, 9, 10, 11, 12, 13, 14, 15);
      v16h bfrag = *(const v16h*)(&sW1[((s * 4 + n) * 32 + lane) * 16]);
      acc = __builtin_amdgcn_wmma_f32_16x16x32_f16(false, afrag, false, bfrag,
                                                   (short)0, acc, false, false);
    }
    const float bias = sLN[128 + n * 16 + lr];
#pragma unroll
    for (int v = 0; v < 8; ++v) {            // D elem (M = v + 8*hi, N = lr)
      float x = acc[v] + bias;
      x = x > 0.f ? x : 0.f;
      sH[w][(v + 8 * hi) * 64 + n * 16 + lr] = (_Float16)x;
    }
  }
  wave_lds_sync();

  // ---- GEMM2: [16x64] x [64x128], bias, store f32 ----
  const _Float16* a2Base = &sH[w][lr * 64];
  float* ob = out + (long long)pbase * EDGE_DIM;
#pragma unroll
  for (int n = 0; n < 8; ++n) {
    v8f acc = {0.f, 0.f, 0.f, 0.f, 0.f, 0.f, 0.f, 0.f};
#pragma unroll
    for (int s = 0; s < 2; ++s) {
      v8h lo  = *(const v8h*)(a2Base + 32 * s + 16 * hi);
      v8h hi8 = *(const v8h*)(a2Base + 32 * s + 16 * hi + 8);
      v16h afrag = __builtin_shufflevector(lo, hi8, 0, 1, 2, 3, 4, 5, 6, 7,
                                           8, 9, 10, 11, 12, 13, 14, 15);
      v16h bfrag = *(const v16h*)(&sW2[((s * 8 + n) * 32 + lane) * 16]);
      acc = __builtin_amdgcn_wmma_f32_16x16x32_f16(false, afrag, false, bfrag,
                                                   (short)0, acc, false, false);
    }
    const float bias = sLN[192 + n * 16 + lr];
#pragma unroll
    for (int v = 0; v < 8; ++v)
      ob[(long long)(v + 8 * hi) * EDGE_DIM + n * 16 + lr] = acc[v] + bias;
  }
}

extern "C" void kernel_launch(void* const* d_in, const int* in_sizes, int n_in,
                              void* d_out, int out_size, void* d_ws, size_t ws_size,
                              hipStream_t stream) {
  const float* x  = (const float*)d_in[0];
  // d_in[1] = z (unused)
  const float* Wi = (const float*)d_in[2];
  const float* bi = (const float*)d_in[3];
  const float* Wj = (const float*)d_in[4];
  const float* bj = (const float*)d_in[5];
  const float* g  = (const float*)d_in[6];
  const float* b  = (const float*)d_in[7];
  const float* W1 = (const float*)d_in[8];
  const float* b1 = (const float*)d_in[9];
  const float* W2 = (const float*)d_in[10];
  const float* b2 = (const float*)d_in[11];

  float* nI = (float*)d_ws;                       // 768*64 f32
  float* nJ = nI + N_NODES * HIDDEN;              // 768*64 f32
  _Float16* w1p = (_Float16*)(nJ + N_NODES * HIDDEN);  // 4096 halves
  _Float16* w2p = w1p + 4096;                          // 8192 halves

  prep_proj<<<N_NODES, 128, 0, stream>>>(x, Wi, bi, Wj, bj, nI, nJ);
  prep_pack<<<1, 256, 0, stream>>>(W1, W2, w1p, w2p);

  const int nblocks = (N_NODES * N_NODES) / 128;  // 128 pairs per block
  edge_main<<<nblocks, 256, 0, stream>>>(nI, nJ, w1p, w2p, g, b, b1, b2,
                                         (float*)d_out);
}
